// Attntopo_46746424049693
// MI455X (gfx1250) — 8-bit matrix operand order repaired and checked by reading, compiled, not run on hardware
//
#include <hip/hip_runtime.h>
#include <hip/hip_bf16.h>
#include <stdint.h>

#define NN_DIM 6144
#define IN_F 256
#define OUT_F 64
#define ALPHA 0.2f
#define NEG_INF_V -9e15f

typedef int v8i __attribute__((ext_vector_type(8)));

__global__ void pack_transpose_kernel(const float* __restrict__ adj,
                                      uint8_t* __restrict__ A8,
                                      uint8_t* __restrict__ AT8) {
    __shared__ uint8_t tile[64][65];
    const int bx = blockIdx.x * 64;
    const int by = blockIdx.y * 64;
    const int t = threadIdx.x;
    const int tx = t & 63;
    const int ty = t >> 6;
    for (int r = 0; r < 16; ++r) {
        const int lr = r * 4 + ty;
        const int row = by + lr;
        const float v = adj[(size_t)row * NN_DIM + bx + tx];
        const uint8_t u = (v > 0.0f) ? (uint8_t)1 : (uint8_t)0;
        A8[(size_t)row * NN_DIM + bx + tx] = u;
        tile[lr][tx] = u;
    }
    __syncthreads();
    for (int r = 0; r < 16; ++r) {
        const int cc = r * 4 + ty;
        AT8[(size_t)(bx + cc) * NN_DIM + by + tx] = tile[tx][cc];
    }
}

__global__ void h_kernel(const float* __restrict__ x, const float* __restrict__ W,
                         float* __restrict__ h) {
    const int t = threadIdx.x;
    const int row = blockIdx.x * 4 + (t >> 6);
    const int c = t & 63;
    float acc = 0.0f;
    const float* xr = x + (size_t)row * IN_F;
    for (int k = 0; k < IN_F; ++k) {
        acc = fmaf(xr[k], W[k * OUT_F + c], acc);
    }
    h[(size_t)row * OUT_F + c] = acc;
}

__global__ void wh_kernel(const float* __restrict__ h, const float* __restrict__ a,
                          float* __restrict__ wh1, float* __restrict__ wh2) {
    const int row = blockIdx.x * 256 + threadIdx.x;
    if (row >= NN_DIM) return;
    const float* hr = h + (size_t)row * OUT_F;
    float a1 = 0.0f, a2 = 0.0f;
    for (int c = 0; c < OUT_F; ++c) {
        const float hv = hr[c];
        a1 = fmaf(hv, a[c], a1);
        a2 = fmaf(hv, a[OUT_F + c], a2);
    }
    wh1[row] = a1;
    wh2[row] = a2;
}

__device__ __forceinline__ v8i load_a_frag(const uint8_t* __restrict__ arow,
                                           int kb, int kh) {
    const uint2 a01 = *(const uint2*)(arow + kb + kh * 8);
    const uint2 a23 = *(const uint2*)(arow + kb + 16 + kh * 8);
    const uint2 a45 = *(const uint2*)(arow + kb + 32 + kh * 8);
    const uint2 a67 = *(const uint2*)(arow + kb + 48 + kh * 8);
    v8i av;
    av[0] = (int)a01.x; av[1] = (int)a01.y; av[2] = (int)a23.x; av[3] = (int)a23.y;
    av[4] = (int)a45.x; av[5] = (int)a45.y; av[6] = (int)a67.x; av[7] = (int)a67.y;
    return av;
}

__device__ __forceinline__ v8i load_b_frag(const uint8_t* __restrict__ brow,
                                           int kb, int kh) {
    const uint2 b01 = *(const uint2*)(brow + kb + kh * 8);
    const uint2 b23 = *(const uint2*)(brow + kb + 16 + kh * 8);
    const uint2 b45 = *(const uint2*)(brow + kb + 32 + kh * 8);
    const uint2 b67 = *(const uint2*)(brow + kb + 48 + kh * 8);
    v8i bv;
    bv[0] = (int)b01.x; bv[1] = (int)b01.y; bv[2] = (int)b23.x; bv[3] = (int)b23.y;
    bv[4] = (int)b45.x; bv[5] = (int)b45.y; bv[6] = (int)b67.x; bv[7] = (int)b67.y;
    return bv;
}

template <int MODE>
__global__ void mm_iu8_kernel(const uint8_t* __restrict__ A,
                              const uint8_t* __restrict__ BT,
                              void* __restrict__ Cout) {
    const int wave = threadIdx.x >> 5;
    const int lane = threadIdx.x & 31;
    const int m0 = (blockIdx.y * 4 + wave) * 32;
    const int n0 = blockIdx.x * 64;
    const int ml = lane & 15;
    const int kh = lane >> 4;

    const uint8_t* arow[2];
    arow[0] = A + (size_t)(m0 + ml) * NN_DIM;
    arow[1] = arow[0] + (size_t)16 * NN_DIM;
    const uint8_t* brow[4];
    #pragma unroll
    for (int nt = 0; nt < 4; ++nt)
        brow[nt] = BT + (size_t)(n0 + nt * 16 + ml) * NN_DIM;

    v8i acc[2][4] = {};
    for (int kb = 0; kb < NN_DIM; kb += 64) {
        v8i av[2], bv[4];
        #pragma unroll
        for (int mt = 0; mt < 2; ++mt) av[mt] = load_a_frag(arow[mt], kb, kh);
        #pragma unroll
        for (int nt = 0; nt < 4; ++nt) bv[nt] = load_b_frag(brow[nt], kb, kh);
        #pragma unroll
        for (int mt = 0; mt < 2; ++mt)
            #pragma unroll
            for (int nt = 0; nt < 4; ++nt)
                acc[mt][nt] = __builtin_amdgcn_wmma_i32_16x16x64_iu8(
                    false, av[mt], false, bv[nt], acc[mt][nt], false, false);
    }

    const int n = lane & 15;
    const int half = lane >> 4;
    #pragma unroll
    for (int mt = 0; mt < 2; ++mt) {
        #pragma unroll
        for (int nt = 0; nt < 4; ++nt) {
            const size_t col = (size_t)(n0 + nt * 16 + n);
            #pragma unroll
            for (int v = 0; v < 8; ++v) {
                const int row = m0 + mt * 16 + v + half * 8;
                int val = acc[mt][nt][v];
                if (MODE == 0) {
                    val = val > 255 ? 255 : (val < 0 ? 0 : val);
                    ((uint8_t*)Cout)[(size_t)row * NN_DIM + col] = (uint8_t)val;
                } else {
                    val = val > 65535 ? 65535 : (val < 0 ? 0 : val);
                    ((uint16_t*)Cout)[(size_t)row * NN_DIM + col] = (uint16_t)val;
                }
            }
        }
    }
}

__global__ void attn_row_kernel(const uint8_t* __restrict__ A8,
                                const uint8_t* __restrict__ A2,
                                const uint16_t* __restrict__ A3,
                                const float* __restrict__ wh1,
                                const float* __restrict__ wh2,
                                const float* __restrict__ h,
                                const float* __restrict__ Wsi,
                                const float* __restrict__ Wei,
                                float* __restrict__ out) {
    __shared__ float sc[NN_DIM];
    __shared__ float red[256];
    const int i = blockIdx.x;
    const int t = threadIdx.x;
    const float wei = fabsf(Wei[0]);
    const float wsi = fabsf(Wsi[0]);
    const float wh1i = wh1[i];
    const size_t rowoff = (size_t)i * NN_DIM;

    float lmax = -INFINITY;
    for (int j = t; j < NN_DIM; j += 256) {
        float s;
        if (A8[rowoff + j]) {
            float e = wh1i + wh2[j];
            e = e > 0.0f ? e : ALPHA * e;
            const float aw = 1.0f + (float)A2[rowoff + j] + (float)A3[rowoff + j];
            s = wei * e + wsi * aw;
        } else {
            s = NEG_INF_V;
        }
        sc[j] = s;
        lmax = fmaxf(lmax, s);
    }
    red[t] = lmax;
    __syncthreads();
    for (int o = 128; o > 0; o >>= 1) {
        if (t < o) red[t] = fmaxf(red[t], red[t + o]);
        __syncthreads();
    }
    const float m = red[0];
    __syncthreads();

    float lsum = 0.0f;
    for (int j = t; j < NN_DIM; j += 256) {
        const float w = expf(sc[j] - m);
        sc[j] = w;
        lsum += w;
    }
    red[t] = lsum;
    __syncthreads();
    for (int o = 128; o > 0; o >>= 1) {
        if (t < o) red[t] += red[t + o];
        __syncthreads();
    }
    const float inv = 1.0f / red[0];
    __syncthreads();

    const int f = t & 63;
    const int part = t >> 6;
    float acc = 0.0f;
    const int j0 = part * (NN_DIM / 4), j1 = j0 + (NN_DIM / 4);
    for (int j = j0; j < j1; ++j) {
        const float w = sc[j];
        if (w != 0.0f) acc = fmaf(w, h[(size_t)j * OUT_F + f], acc);
    }
    red[t] = acc;
    __syncthreads();
    if (part == 0) {
        float v = (red[f] + red[64 + f] + red[128 + f] + red[192 + f]) * inv;
        v = v > 0.0f ? v : expm1f(v);
        out[(size_t)i * OUT_F + f] = v;
    }
}

extern "C" void kernel_launch(void* const* d_in, const int* in_sizes, int n_in,
                              void* d_out, int out_size, void* d_ws, size_t ws_size,
                              hipStream_t stream) {
    (void)in_sizes; (void)n_in; (void)out_size; (void)ws_size;
    const float* x   = (const float*)d_in[0];
    const float* adj = (const float*)d_in[1];
    const float* W   = (const float*)d_in[2];
    const float* a   = (const float*)d_in[3];
    const float* Wsi = (const float*)d_in[4];
    const float* Wei = (const float*)d_in[5];
    float* out = (float*)d_out;

    const size_t NN = (size_t)NN_DIM * NN_DIM;
    uint8_t*  adj8  = (uint8_t*)d_ws;
    uint8_t*  adjT8 = adj8 + NN;
    uint8_t*  adj2  = adjT8 + NN;
    uint16_t* adj3  = (uint16_t*)(adj2 + NN);
    float*    hbuf  = (float*)((uint8_t*)d_ws + 5 * NN);
    float*    wh1   = hbuf + (size_t)NN_DIM * OUT_F;
    float*    wh2   = wh1 + NN_DIM;

    pack_transpose_kernel<<<dim3(NN_DIM / 64, NN_DIM / 64), 256, 0, stream>>>(adj, adj8, adjT8);
    h_kernel<<<NN_DIM / 4, 256, 0, stream>>>(x, W, hbuf);
    wh_kernel<<<NN_DIM / 256, 256, 0, stream>>>(hbuf, a, wh1, wh2);
    mm_iu8_kernel<0><<<dim3(NN_DIM / 64, NN_DIM / 128), 128, 0, stream>>>(adj8, adjT8, (void*)adj2);
    mm_iu8_kernel<1><<<dim3(NN_DIM / 64, NN_DIM / 128), 128, 0, stream>>>(adj2, adjT8, (void*)adj3);
    attn_row_kernel<<<NN_DIM, 256, 0, stream>>>(adj8, adj2, adj3, wh1, wh2, hbuf,
                                                Wsi, Wei, out);
}
